// GRUCell_43181601194444
// MI455X (gfx1250) — matrix operand order repaired and checked by reading, not compiled, not run on hardware
//
#include <hip/hip_runtime.h>

typedef __attribute__((ext_vector_type(16))) _Float16 v16h;
typedef __attribute__((ext_vector_type(8)))  _Float16 v8h;
typedef __attribute__((ext_vector_type(4)))  _Float16 v4h;
typedef __attribute__((ext_vector_type(8)))  float    v8f;

#define B_ROWS   16384
#define KDIM     256
#define NB_COLS  768      // 3H
#define M_TILE   32       // rows per workgroup
#define LDSP     264      // padded LDS row stride in halves (528B = 33*16 -> 16B align kept)
#define NT_G1    6        // n-tiles per wave, GEMM1 (48 tiles / 8 waves)
#define NT_G2    2        // n-tiles per wave, GEMM2 (16 tiles / 8 waves)
#define KSTEPS   8        // 256 / 32
#define REG_SZ   (M_TILE * LDSP)   // halves per LDS region

// ---- pre-swizzled f16 weight blocks in workspace -------------------------
// block(m, t, s) = 32x16 (KxN) f16 tile in contiguous-k B lane order (same k order as load_a_lds):
//   lane L, half h -> K = s*32 + (L&16) + h, N = t*16 + (L&15)
// halves offset = ((m*48 + t)*8 + s)*512 + L*16   (1 KB per block, 768 blocks)
__device__ __forceinline__ const _Float16* b_block(const _Float16* ws, int m, int t, int s, int lane) {
    return ws + (((m * 48 + t) * 8 + s) * 512) + lane * 16;
}

__global__ void gru_prep_weights(const float* __restrict__ W,
                                 const float* __restrict__ Wh,
                                 _Float16* __restrict__ ws) {
    const int bid  = blockIdx.x;        // 0..767
    const int lane = threadIdx.x;       // 0..31
    const int m = bid / 384;
    const int r = bid - m * 384;
    const int t = r >> 3;
    const int s = r & 7;
    const float* src = m ? Wh : W;
    const int N     = t * 16 + (lane & 15);
    const int Kbase = s * 32 + (lane & 16);
    _Float16* dst = ws + bid * 512 + lane * 16;
#pragma unroll
    for (int h = 0; h < 16; ++h)
        dst[h] = (_Float16)src[(Kbase + h) * NB_COLS + N];
}

// ---- A-operand (16x32 f16, MxK) loader from LDS --------------------------
// lane L: M = L&15; halves 0..15 -> K = K0..K0+15
// where K0 = s*32 + ((L>>4)&1)*16
__device__ __forceinline__ v16h load_a_lds(const _Float16* lds_row, int K0) {
    v8h lo = *(const v8h*)(lds_row + K0);
    v8h hi = *(const v8h*)(lds_row + K0 + 8);
    return __builtin_shufflevector(lo, hi, 0,1,2,3,4,5,6,7,8,9,10,11,12,13,14,15);
}

__device__ __forceinline__ float sigmoidf_fast(float x) { return 1.0f / (1.0f + __expf(-x)); }
__device__ __forceinline__ float tanhf_fast(float x)    { return 1.0f - 2.0f / (__expf(2.0f * x) + 1.0f); }

__global__ void __launch_bounds__(256)
gru_main(const float* __restrict__ x,
         const float* __restrict__ h,
         const float* __restrict__ bias,
         const _Float16* __restrict__ ws,
         float* __restrict__ out) {
    // Phase-aliased LDS: R0 = x_f16 then r*h ; R1 = h_f16 then u ; R2 = xc
    __shared__ __align__(16) _Float16 smem[3 * REG_SZ];
    _Float16* R0 = smem;
    _Float16* R1 = smem + REG_SZ;
    _Float16* R2 = smem + 2 * REG_SZ;

    const int lane  = threadIdx.x & 31;
    const int wavei = __builtin_amdgcn_readfirstlane((int)(threadIdx.x >> 5)); // scalar wave id
    const int row0  = blockIdx.x * M_TILE;
    const int mrow  = lane & 15;
    const int hi16  = (lane >> 4) & 1;

    // ---- stage x, h -> LDS f16 (cooperative, one convert per element) ----
    {
        const int tid = threadIdx.x;
#pragma unroll
        for (int it = 0; it < 8; ++it) {
            const int idx  = it * 256 + tid;     // float4 index within 32x256 tile
            const int row  = idx >> 6;           // 64 float4 per row
            const int col4 = (idx & 63) * 4;
            const float4 fx = *(const float4*)(x + (size_t)(row0 + row) * KDIM + col4);
            const float4 fh = *(const float4*)(h + (size_t)(row0 + row) * KDIM + col4);
            v4h vx, vh;
            vx[0] = (_Float16)fx.x; vx[1] = (_Float16)fx.y; vx[2] = (_Float16)fx.z; vx[3] = (_Float16)fx.w;
            vh[0] = (_Float16)fh.x; vh[1] = (_Float16)fh.y; vh[2] = (_Float16)fh.z; vh[3] = (_Float16)fh.w;
            *(v4h*)(R0 + row * LDSP + col4) = vx;
            *(v4h*)(R1 + row * LDSP + col4) = vh;
        }
    }
    __syncthreads();

    // ================= GEMM1: acc = x@W  (+ h@Wh for cols < 512) ==========
    v8f acc[2][NT_G1];
#pragma unroll
    for (int mt = 0; mt < 2; ++mt)
#pragma unroll
        for (int nt = 0; nt < NT_G1; ++nt)
#pragma unroll
            for (int e = 0; e < 8; ++e) acc[mt][nt][e] = 0.0f;

#pragma unroll
    for (int s = 0; s < KSTEPS; ++s) {
        const int K0 = s * 32 + (hi16 << 4);
        v16h ax[2], ah[2];
#pragma unroll
        for (int mt = 0; mt < 2; ++mt) {
            const int arow = (mt * 16 + mrow) * LDSP;
            ax[mt] = load_a_lds(R0 + arow, K0);
            ah[mt] = load_a_lds(R1 + arow, K0);
        }
#pragma unroll
        for (int nt = 0; nt < NT_G1; ++nt) {
            const int gnt = wavei * NT_G1 + nt;     // scalar: global 16-col tile index
            v16h bw = *(const v16h*)b_block(ws, 0, gnt, s, lane);
#pragma unroll
            for (int mt = 0; mt < 2; ++mt)
                acc[mt][nt] = __builtin_amdgcn_wmma_f32_16x16x32_f16(
                    false, ax[mt], false, bw, (short)0, acc[mt][nt], false, false);
            if (gnt < 32) {                          // r,u columns also get h@Wh
                v16h bh = *(const v16h*)b_block(ws, 1, gnt, s, lane);
#pragma unroll
                for (int mt = 0; mt < 2; ++mt)
                    acc[mt][nt] = __builtin_amdgcn_wmma_f32_16x16x32_f16(
                        false, ah[mt], false, bh, (short)0, acc[mt][nt], false, false);
            }
        }
    }
    __syncthreads();   // all waves done reading x_f16/h_f16 before overwrite

    // ------- GEMM1 epilogue: gates -> LDS (R0 := r*h, R1 := u, R2 := xc) --
    const float* hbase = h + (size_t)row0 * KDIM;
#pragma unroll
    for (int nt = 0; nt < NT_G1; ++nt) {
        const int gcol = wavei * 96 + nt * 16 + mrow;   // 0..767
        const float bv = bias[gcol];
#pragma unroll
        for (int mt = 0; mt < 2; ++mt) {
#pragma unroll
            for (int r = 0; r < 8; ++r) {
                const int row_l = mt * 16 + hi16 * 8 + r;   // D layout: M = r + 8*hi16
                const float v = acc[mt][nt][r] + bv;
                if (gcol < 256) {
                    const float rg = sigmoidf_fast(v);
                    const float hv = hbase[row_l * KDIM + gcol];   // fp32 h from L1/L2
                    R0[row_l * LDSP + gcol] = (_Float16)(rg * hv);
                } else if (gcol < 512) {
                    R1[row_l * LDSP + (gcol - 256)] = (_Float16)sigmoidf_fast(v);
                } else {
                    R2[row_l * LDSP + (gcol - 512)] = (_Float16)v;
                }
            }
        }
    }
    __syncthreads();

    // ================= GEMM2: cand_pre = (r*h) @ Wh_c =====================
    v8f acc2[2][NT_G2];
#pragma unroll
    for (int mt = 0; mt < 2; ++mt)
#pragma unroll
        for (int j = 0; j < NT_G2; ++j)
#pragma unroll
            for (int e = 0; e < 8; ++e) acc2[mt][j][e] = 0.0f;

#pragma unroll
    for (int s = 0; s < KSTEPS; ++s) {
        const int K0 = s * 32 + (hi16 << 4);
        v16h ar[2];
#pragma unroll
        for (int mt = 0; mt < 2; ++mt)
            ar[mt] = load_a_lds(R0 + (mt * 16 + mrow) * LDSP, K0);
#pragma unroll
        for (int j = 0; j < NT_G2; ++j) {
            v16h bh = *(const v16h*)b_block(ws, 1, 32 + wavei * NT_G2 + j, s, lane);
#pragma unroll
            for (int mt = 0; mt < 2; ++mt)
                acc2[mt][j] = __builtin_amdgcn_wmma_f32_16x16x32_f16(
                    false, ar[mt], false, bh, (short)0, acc2[mt][j], false, false);
        }
    }

    // ------- fuse gates, write h_t ---------------------------------------
#pragma unroll
    for (int j = 0; j < NT_G2; ++j) {
        const int c = wavei * 32 + j * 16 + mrow;       // 0..255
#pragma unroll
        for (int mt = 0; mt < 2; ++mt) {
#pragma unroll
            for (int r = 0; r < 8; ++r) {
                const int row_l = mt * 16 + hi16 * 8 + r;
                const float pre  = acc2[mt][j][r] + (float)R2[row_l * LDSP + c];
                const float cand = tanhf_fast(pre);
                const float u    = (float)R1[row_l * LDSP + c];
                const float hv   = hbase[row_l * KDIM + c];
                out[(size_t)(row0 + row_l) * KDIM + c] = u * hv + (1.0f - u) * cand;
            }
        }
    }
}

extern "C" void kernel_launch(void* const* d_in, const int* in_sizes, int n_in,
                              void* d_out, int out_size, void* d_ws, size_t ws_size,
                              hipStream_t stream) {
    const float* x    = (const float*)d_in[0];
    const float* h    = (const float*)d_in[1];
    const float* W    = (const float*)d_in[2];   // [256, 768]
    const float* Wh   = (const float*)d_in[3];   // [256, 768]
    const float* bias = (const float*)d_in[4];   // [768]
    _Float16* ws = (_Float16*)d_ws;              // needs 768 KB

    gru_prep_weights<<<dim3(2 * 48 * 8), dim3(32), 0, stream>>>(W, Wh, ws);
    gru_main<<<dim3(B_ROWS / M_TILE), dim3(256), 0, stream>>>(x, h, bias, ws, (float*)d_out);
}
